// StateActionGATCriticMultiHead_63548336111654
// MI455X (gfx1250) — compile-verified, matrix operand order checked by reading, not run on hardware
//
#include <hip/hip_runtime.h>
#include <hip/hip_bf16.h>

typedef _Float16 v16h __attribute__((ext_vector_type(16)));
typedef float    v8f  __attribute__((ext_vector_type(8)));

#define B_      128
#define N_      32
#define DO_     128
#define NA_     16
#define H_      4
#define DK_     64
#define DV_     64
#define DM_     128   // d_model after embed
#define F1IN_   256   // H*Dv
#define F1OUT_  64
#define MH_STR  264   // padded LDS strides (halfs) to dodge bank conflicts
#define W1_STR  264
#define D_STR   136   // stride for [*][128] f16 tiles
#define QK_STR  72    // stride for [*][64] f16 tiles
#define T_STR   40    // stride for [*][32] f16 tiles

__device__ __forceinline__ float leakyf(float x) { return x > 0.f ? x : 0.01f * x; }

// One 16x16 output tile, K = 32*ksteps, f16 A/B rows contiguous in k.
// aRow points at A's tile-row (m16), bRow at B's tile-column (m16); lh = lane>>4.
__device__ __forceinline__ v8f wmma_tile(const _Float16* aRow, const _Float16* bRow,
                                         int ksteps, int lh, v8f acc)
{
    for (int kk = 0; kk < ksteps; ++kk) {
        const int k0 = kk * 32 + lh * 8;   // ISA f16 A/B K-striping
        v16h av, bv;
#pragma unroll
        for (int e = 0; e < 8; ++e) {
            av[e]     = aRow[k0 + e];
            av[e + 8] = aRow[k0 + 16 + e];
            bv[e]     = bRow[k0 + e];
            bv[e + 8] = bRow[k0 + 16 + e];
        }
        acc = __builtin_amdgcn_wmma_f32_16x16x32_f16(false, av, false, bv,
                                                     (short)0, acc, false, false);
    }
    return acc;
}

// ---------------------------------------------------------------------------
// Kernel 0: one-time weight repack to f16 (transposed, column-contiguous-in-k)
//   w1t_h  [64][256]   = W_f1^T
//   wqkT_h [H][128][128]: col c<64 -> Wq[h][:, c], c>=64 -> Wk[h][:, c-64]
//   wvT_h  [H][64][128] = Wv[h]^T
// ---------------------------------------------------------------------------
__global__ __launch_bounds__(256) void k_prep(
    const float* __restrict__ Wq, const float* __restrict__ Wk,
    const float* __restrict__ Wv, const float* __restrict__ W_f1,
    _Float16* __restrict__ w1t_h, _Float16* __restrict__ wqkT_h,
    _Float16* __restrict__ wvT_h)
{
    const int gid = blockIdx.x * 256 + threadIdx.x;   // 16384 threads
    for (int t = gid; t < F1OUT_ * F1IN_; t += 16384) {
        const int n = t >> 8, k = t & 255;
        w1t_h[t] = (_Float16)W_f1[k * F1OUT_ + n];
    }
    for (int t = gid; t < H_ * 128 * 128; t += 16384) {
        const int h = t >> 14, c = (t >> 7) & 127, d = t & 127;
        const float v = (c < 64) ? Wq[(h * DM_ + d) * DK_ + c]
                                 : Wk[(h * DM_ + d) * DK_ + (c - 64)];
        wqkT_h[t] = (_Float16)v;
    }
    for (int t = gid; t < H_ * 64 * 128; t += 16384) {
        const int h = t >> 13, e = (t >> 7) & 63, d = t & 127;
        wvT_h[t] = (_Float16)Wv[(h * DM_ + d) * DV_ + e];
    }
}

// ---------------------------------------------------------------------------
// Kernel 1: per-batch embeddings (scalar fp32; streams each input once).
// ---------------------------------------------------------------------------
__global__ __launch_bounds__(256) void k_embed(
    const float* __restrict__ states, const float* __restrict__ policies,
    const float* __restrict__ actions,
    const float* __restrict__ W_se, const float* __restrict__ b_se,
    const float* __restrict__ W_sape, const float* __restrict__ b_sape,
    float* __restrict__ se_g, float* __restrict__ xa_g, float* __restrict__ xp_g)
{
    __shared__ float stL[N_ * DO_];
    __shared__ float actL[N_ * NA_];
    __shared__ float polL[N_ * NA_];
    const int b = blockIdx.x, tid = threadIdx.x;

    for (int t = tid; t < N_ * DO_; t += 256) stL[t] = states[b * N_ * DO_ + t];
    for (int t = tid; t < N_ * NA_; t += 256) {
        actL[t] = actions[b * N_ * NA_ + t];
        polL[t] = policies[b * N_ * NA_ + t];
    }
    __syncthreads();

    for (int o = tid; o < N_ * DM_; o += 256) {
        const int n = o >> 7, c = o & 127;
        float acc = b_se[c];
        for (int k = 0; k < DO_; ++k) acc += stL[n * DO_ + k] * W_se[k * DM_ + c];
        se_g[b * N_ * DM_ + o] = leakyf(acc);
    }
    for (int o = tid; o < N_ * DM_; o += 256) {
        const int n = o >> 7, c = o & 127;
        float s = b_sape[c];
        for (int k = 0; k < DO_; ++k) s += stL[n * DO_ + k] * W_sape[k * DM_ + c];
        float a = s, p = s;
        for (int k = 0; k < NA_; ++k) {
            const float wv = W_sape[(DO_ + k) * DM_ + c];
            a += actL[n * NA_ + k] * wv;
            p += polL[n * NA_ + k] * wv;
        }
        xa_g[b * N_ * DM_ + o] = leakyf(a);
        xp_g[b * N_ * DM_ + o] = leakyf(p);
    }
}

// ---------------------------------------------------------------------------
// Kernel 2: per-(b,h) Q/K projection + scores + softmax, all GEMMs on WMMA.
//   GEMM1: [32 x 128] se_f16 @ [128 x 128] (Wq|Wk)  -> Q,K rows (f16 LDS)
//   GEMM2: scores = Q @ K^T  (K row-major == B column-major requirement)
// ---------------------------------------------------------------------------
__global__ __launch_bounds__(256) void k_qk(
    const float* __restrict__ se_g, const _Float16* __restrict__ wqkT_h,
    float* __restrict__ w_out)
{
    __shared__ __align__(16) _Float16 seH[N_ * D_STR];      //  8.7 KB
    __shared__ __align__(16) _Float16 wqkT[128 * D_STR];    // 34.8 KB
    __shared__ __align__(16) _Float16 qLh[N_ * QK_STR];     //  4.6 KB
    __shared__ __align__(16) _Float16 kLh[N_ * QK_STR];     //  4.6 KB
    __shared__ float scoresL[N_ * 36];                      //  4.6 KB
    const int blk = blockIdx.x, b = blk >> 2, h = blk & 3;
    const int tid = threadIdx.x;
    const int lane = tid & 31, wv = tid >> 5;
    const int lh = lane >> 4, m16 = lane & 15;

    // stage se (f32->f16) and pre-packed head weights (f16 copy, contiguous)
    for (int u = 0; u < 16; ++u) {
        const int t = tid * 16 + u;
        seH[(t >> 7) * D_STR + (t & 127)] = (_Float16)se_g[b * N_ * DM_ + t];
    }
    {
        const _Float16* src = wqkT_h + h * 128 * 128;
        for (int u = 0; u < 64; ++u) {
            const int t = tid * 64 + u;
            wqkT[(t >> 7) * D_STR + (t & 127)] = src[t];
        }
    }
    __syncthreads();

    // GEMM1: 16 tiles (2 row-blocks x 8 col-blocks), 2 per wave, K=128
    for (int t2 = 0; t2 < 2; ++t2) {
        const int id = wv * 2 + t2;
        const int jb = id >> 3, cb = id & 7;
        v8f acc = {0.f, 0.f, 0.f, 0.f, 0.f, 0.f, 0.f, 0.f};
        acc = wmma_tile(seH + (jb * 16 + m16) * D_STR,
                        wqkT + (cb * 16 + m16) * D_STR, 4, lh, acc);
        const int col = cb * 16 + m16;     // tile-uniform branch (cb decides)
#pragma unroll
        for (int r = 0; r < 8; ++r) {
            const int row = jb * 16 + lh * 8 + r;
            if (col < 64) qLh[row * QK_STR + col] = (_Float16)acc[r];
            else          kLh[row * QK_STR + (col - 64)] = (_Float16)acc[r];
        }
    }
    __syncthreads();

    // GEMM2: scores[32x32] = Q @ K^T, 4 tiles on waves 0..3, K=64
    if (wv < 4) {
        const int ib = wv >> 1, jbb = wv & 1;
        v8f acc = {0.f, 0.f, 0.f, 0.f, 0.f, 0.f, 0.f, 0.f};
        acc = wmma_tile(qLh + (ib * 16 + m16) * QK_STR,
                        kLh + (jbb * 16 + m16) * QK_STR, 2, lh, acc);
#pragma unroll
        for (int r = 0; r < 8; ++r)
            scoresL[(ib * 16 + lh * 8 + r) * 36 + jbb * 16 + m16] = acc[r] * 0.125f;
    }
    __syncthreads();

    // softmax (one thread per row), write w to its d_out slot
    if (tid < N_) {
        const int i = tid;
        float sc[N_];
        float mx = -1e30f;
        for (int j = 0; j < N_; ++j) { sc[j] = scoresL[i * 36 + j]; mx = sc[j] > mx ? sc[j] : mx; }
        float sum = 0.f;
        for (int j = 0; j < N_; ++j) { sc[j] = __expf(sc[j] - mx); sum += sc[j]; }
        const float inv = 1.f / sum;
        for (int j = 0; j < N_; ++j)
            w_out[((b * H_ + h) * N_ + i) * N_ + j] = sc[j] * inv;
    }
}

// ---------------------------------------------------------------------------
// Kernel 3: per-(b,h) value path on WMMA.
//   GEMM3: [xa;xp] (64x128) @ Wv (128x64) -> avaT/avpT (f16, B-layout)
//   delta = avp - ava  (fp32 to ws)
//   GEMM4: base = w (32x32, f16) @ ava (32x64), K=32 single step
// ---------------------------------------------------------------------------
__global__ __launch_bounds__(256) void k_av(
    const float* __restrict__ xa_g, const float* __restrict__ xp_g,
    const _Float16* __restrict__ wvT_h, const float* __restrict__ w_g,
    float* __restrict__ base_g, float* __restrict__ delta_g)
{
    __shared__ __align__(16) _Float16 xaxpH[64 * D_STR];    // 17.4 KB
    __shared__ __align__(16) _Float16 wvTH[64 * D_STR];     // 17.4 KB
    __shared__ __align__(16) _Float16 avaTh[64 * T_STR];    //  5.1 KB  [e][j]
    __shared__ __align__(16) _Float16 avpTh[64 * T_STR];    //  5.1 KB
    __shared__ __align__(16) _Float16 wLh[N_ * T_STR];      //  2.5 KB
    const int blk = blockIdx.x, b = blk >> 2, h = blk & 3;
    const int tid = threadIdx.x;
    const int lane = tid & 31, wv = tid >> 5;
    const int lh = lane >> 4, m16 = lane & 15;

    for (int u = 0; u < 32; ++u) {   // rows 0..31 = xa, 32..63 = xp (f32->f16)
        const int t = tid * 32 + u;
        const int row = t >> 7, d = t & 127;
        const float v = (row < 32) ? xa_g[(b * N_ + row) * DM_ + d]
                                   : xp_g[(b * N_ + (row - 32)) * DM_ + d];
        xaxpH[row * D_STR + d] = (_Float16)v;
    }
    {
        const _Float16* src = wvT_h + h * 64 * 128;
        for (int u = 0; u < 32; ++u) {
            const int t = tid * 32 + u;
            wvTH[(t >> 7) * D_STR + (t & 127)] = src[t];
        }
    }
    for (int u = 0; u < 4; ++u) {    // attention weights (already in d_out)
        const int t = tid * 4 + u;
        wLh[(t >> 5) * T_STR + (t & 31)] =
            (_Float16)w_g[((b * H_ + h) * N_ + (t >> 5)) * N_ + (t & 31)];
    }
    __syncthreads();

    // GEMM3: 16 tiles (4 row-blocks x 4 col-blocks), 2 per wave, K=128
    for (int t2 = 0; t2 < 2; ++t2) {
        const int id = wv * 2 + t2;
        const int jb = id >> 2, cb = id & 3;
        v8f acc = {0.f, 0.f, 0.f, 0.f, 0.f, 0.f, 0.f, 0.f};
        acc = wmma_tile(xaxpH + (jb * 16 + m16) * D_STR,
                        wvTH + (cb * 16 + m16) * D_STR, 4, lh, acc);
        const int e = cb * 16 + m16;
#pragma unroll
        for (int r = 0; r < 8; ++r) {   // transpose-store into B-layout [e][j]
            const int row = jb * 16 + lh * 8 + r;   // tile-uniform jb decides
            if (row < 32) avaTh[e * T_STR + row] = (_Float16)acc[r];
            else          avpTh[e * T_STR + (row - 32)] = (_Float16)acc[r];
        }
    }
    __syncthreads();

    // delta = avp - ava (fp32)
    for (int u = 0; u < 8; ++u) {
        const int t = tid * 8 + u;
        const int j = t >> 6, e = t & 63;
        delta_g[((b * H_ + h) * N_ + j) * DV_ + e] =
            (float)avpTh[e * T_STR + j] - (float)avaTh[e * T_STR + j];
    }

    // GEMM4: base[32x64] = w @ ava, 8 tiles (2x4), one K=32 step per wave
    {
        const int rb = wv >> 2, cbb = wv & 3;
        v8f acc = {0.f, 0.f, 0.f, 0.f, 0.f, 0.f, 0.f, 0.f};
        acc = wmma_tile(wLh + (rb * 16 + m16) * T_STR,
                        avaTh + (cbb * 16 + m16) * T_STR, 1, lh, acc);
#pragma unroll
        for (int r = 0; r < 8; ++r)
            base_g[((b * H_ + h) * N_ + rb * 16 + lh * 8 + r) * DV_ + cbb * 16 + m16] = acc[r];
    }
}

// ---------------------------------------------------------------------------
// Kernel 4: per-(b,i) fused layernorm + head-concat + FFN via WMMA.
// ---------------------------------------------------------------------------
__global__ __launch_bounds__(256) void k_ffn(
    const float* __restrict__ w_g, const float* __restrict__ base_g,
    const float* __restrict__ delta_g,
    const float* __restrict__ ln_g, const float* __restrict__ ln_b,
    const _Float16* __restrict__ w1t_h, const float* __restrict__ W_f2,
    float* __restrict__ val_out)
{
    __shared__ __align__(16) _Float16 mhL[N_ * MH_STR];       // ~16.9 KB
    __shared__ __align__(16) _Float16 w1L[F1OUT_ * W1_STR];   // ~33.8 KB
    __shared__ float wRow[H_ * N_];
    __shared__ float baseI[F1IN_];
    __shared__ float w2L[F1OUT_];
    __shared__ float partL[4 * N_];
    const int blk = blockIdx.x, b = blk >> 5, i = blk & 31;
    const int tid = threadIdx.x;

    // stage pre-packed f16 W_f1^T (contiguous 128B per thread)
    for (int u = 0; u < 64; ++u) {
        const int t = tid * 64 + u;
        w1L[(t >> 8) * W1_STR + (t & 255)] = w1t_h[t];
    }
    if (tid < F1OUT_) w2L[tid] = W_f2[tid];
    if (tid < H_ * N_)
        wRow[tid] = w_g[((b * H_ + (tid >> 5)) * N_ + i) * N_ + (tid & 31)];
    if (tid < F1IN_)
        baseI[tid] = base_g[((b * H_ + (tid >> 6)) * N_ + i) * DV_ + (tid & 63)];
    __syncthreads();

    // nf + layernorm -> f16 mh tile (one thread per (h,j) row)
    if (tid < H_ * N_) {
        const int h = tid >> 5, j = tid & 31;
        const float wij = wRow[h * N_ + j];
        const float* dG = delta_g + ((b * H_ + h) * N_ + j) * DV_;
        const float* bI = baseI + h * DV_;
        float nf[DV_];
        float mu = 0.f;
        for (int e = 0; e < DV_; ++e) { const float v = bI[e] + wij * dG[e]; nf[e] = v; mu += v; }
        mu *= (1.f / DV_);
        float var = 0.f;
        for (int e = 0; e < DV_; ++e) { const float d = nf[e] - mu; var += d * d; }
        var *= (1.f / DV_);
        const float inv = rsqrtf(var + 1e-5f);
        for (int e = 0; e < DV_; ++e) {
            const float y = (nf[e] - mu) * inv * ln_g[h * DV_ + e] + ln_b[h * DV_ + e];
            mhL[j * MH_STR + h * DV_ + e] = (_Float16)y;
        }
    }
    __syncthreads();

    // WMMA: hidden[32 x 64] = mh[32 x 256] @ W_f1[256 x 64], one tile per wave
    const int lane = tid & 31, wv = tid >> 5;
    const int jb = wv >> 2, cb = wv & 3;
    const int lh = lane >> 4, m16 = lane & 15;
    v8f acc = {0.f, 0.f, 0.f, 0.f, 0.f, 0.f, 0.f, 0.f};
    acc = wmma_tile(mhL + (jb * 16 + m16) * MH_STR,
                    w1L + (cb * 16 + m16) * W1_STR, 8, lh, acc);

    // leaky + dot with W_f2; deterministic shuffle + LDS reduction
    const float w2n = w2L[cb * 16 + m16];
#pragma unroll
    for (int r = 0; r < 8; ++r) {
        float hv = acc[r];
        hv = hv > 0.f ? hv : 0.01f * hv;
        float p = hv * w2n;
        p += __shfl_xor(p, 1, 16);
        p += __shfl_xor(p, 2, 16);
        p += __shfl_xor(p, 4, 16);
        p += __shfl_xor(p, 8, 16);
        if (m16 == 0) partL[cb * N_ + jb * 16 + lh * 8 + r] = p;
    }
    __syncthreads();
    if (tid < N_) {
        const float v = partL[tid] + partL[N_ + tid] + partL[2 * N_ + tid] + partL[3 * N_ + tid];
        val_out[(b * N_ + i) * N_ + tid] = v;
    }
}

// ---------------------------------------------------------------------------
extern "C" void kernel_launch(void* const* d_in, const int* in_sizes, int n_in,
                              void* d_out, int out_size, void* d_ws, size_t ws_size,
                              hipStream_t stream)
{
    const float* states   = (const float*)d_in[0];
    const float* policies = (const float*)d_in[1];
    const float* actions  = (const float*)d_in[2];
    const float* W_se     = (const float*)d_in[3];
    const float* b_se     = (const float*)d_in[4];
    const float* W_sape   = (const float*)d_in[5];
    const float* b_sape   = (const float*)d_in[6];
    const float* Wk       = (const float*)d_in[7];
    const float* Wq       = (const float*)d_in[8];
    const float* Wv       = (const float*)d_in[9];
    const float* ln_g     = (const float*)d_in[10];
    const float* ln_b     = (const float*)d_in[11];
    const float* W_f1     = (const float*)d_in[12];
    const float* W_f2     = (const float*)d_in[13];

    float* val_out = (float*)d_out;                 // [B,N,N,1] = 131072
    float* w_out   = (float*)d_out + B_ * N_ * N_;  // [B,H,N,N] = 524288

    float* ws      = (float*)d_ws;
    float* se_g    = ws;                        // B*N*128 = 524288 floats
    float* xa_g    = ws + 524288;
    float* xp_g    = ws + 1048576;
    float* base_g  = ws + 1572864;              // B*H*N*64 = 1048576 floats
    float* delta_g = ws + 2621440;              // fp32 region ends @ 3670016
    _Float16* h_base = (_Float16*)(ws + 3670016);
    _Float16* w1t_h  = h_base;                  // 16384 halfs
    _Float16* wqkT_h = h_base + 16384;          // 65536 halfs
    _Float16* wvT_h  = h_base + 81920;          // 32768 halfs (~14.9 MiB total)

    k_prep<<<64, 256, 0, stream>>>(Wq, Wk, Wv, W_f1, w1t_h, wqkT_h, wvT_h);
    k_embed<<<B_, 256, 0, stream>>>(states, policies, actions,
                                    W_se, b_se, W_sape, b_sape,
                                    se_g, xa_g, xp_g);
    k_qk<<<B_ * H_, 256, 0, stream>>>(se_g, wqkT_h, w_out);
    k_av<<<B_ * H_, 256, 0, stream>>>(xa_g, xp_g, wvT_h, w_out, base_g, delta_g);
    k_ffn<<<B_ * N_, 256, 0, stream>>>(w_out, base_g, delta_g,
                                       ln_g, ln_b, w1t_h, W_f2, val_out);
}
